// MarkovTransitionModule_2190433321442
// MI455X (gfx1250) — matrix operand order repaired and checked by reading, not compiled, not run on hardware
//
#include <hip/hip_runtime.h>
#include <hip/hip_bf16.h>

// ---------------------------------------------------------------------------
// MarkovTransitionModule for MI455X (gfx1250, wave32, WMMA bf16 + TDM path)
// B=32, L=1024, D=1024, S=32, TAU=1.0
// ---------------------------------------------------------------------------

typedef __attribute__((ext_vector_type(16))) __bf16 v16bf;
typedef __attribute__((ext_vector_type(8)))  __bf16 v8bf;
typedef __attribute__((ext_vector_type(4)))  __bf16 v4bf;
typedef __attribute__((ext_vector_type(8)))  float  v8f;
typedef __attribute__((ext_vector_type(4)))  unsigned int v4u;
typedef __attribute__((ext_vector_type(8)))  int    v8i;
typedef __attribute__((ext_vector_type(4)))  int    v4i;

#define DEV static __device__ __forceinline__

constexpr int Bv = 32, Lv = 1024, Dv = 1024, Sv = 32;
constexpr int M1 = Bv * Lv;        // 32768 rows (B*L)
constexpr int M2 = Bv * (Lv - 1);  // 32736 rows (B*(L-1))

#if defined(__HIP_DEVICE_COMPILE__) && __has_builtin(__builtin_amdgcn_tensor_load_to_lds)
#define USE_TDM 1
#else
#define USE_TDM 0
#endif

// D = A(16x32 bf16) * B(32x16 bf16) + C(16x16 f32)
DEV v8f wmma_bf16(v16bf a, v16bf b, v8f c) {
  return __builtin_amdgcn_wmma_f32_16x16x32_bf16(
      false, a, false, b, (short)0, c, false, false);
}

// A fragment 16x32 from row-major bf16 LDS tile. Lane&15 = row, lane>=16
// selects K 16..31 (contiguous). Two 16B ds_load_b128 per lane.
DEV v16bf load_a16(const __bf16* base, int stride, int lane) {
  const __bf16* p = base + (lane & 15) * stride + ((lane >> 4) << 4);
  v8bf lo = *reinterpret_cast<const v8bf*>(p);
  v8bf hi = *reinterpret_cast<const v8bf*>(p + 8);
  return __builtin_shufflevector(lo, hi, 0, 1, 2, 3, 4, 5, 6, 7,
                                 8, 9, 10, 11, 12, 13, 14, 15);
}

// B fragment from fragment-major (swizzled) LDS: [frag][lane][16 bf16].
// One 32B vector load per lane (2x ds_load_b128), zero repacking VALU.
DEV v16bf load_b_sw(const __bf16* sw, int frag, int lane) {
  return *reinterpret_cast<const v16bf*>(sw + frag * 512 + lane * 16);
}

// Stage a 32(K) x NC(cols) fp32 tile into fragment-major bf16 LDS layout.
// Element (k,n) -> sw[(n>>4)*512 + (((n&15)|(k&16))<<4) + (k&15)].
// Each thread packs 4 consecutive K values -> one 8B ds_store_b64.
template <int NC>
DEV void stage_b_sw(__bf16* sw, const float* __restrict__ src, int srcStride,
                    int tid) {
#pragma unroll
  for (int g = tid; g < NC * 8; g += 256) {
    const int n  = g & (NC - 1);
    const int kq = (g / NC) * 4;
    v4bf pk;
#pragma unroll
    for (int j = 0; j < 4; ++j)
      pk[j] = (__bf16)src[(size_t)(kq + j) * srcStride + n];
    __bf16* d = sw + (n >> 4) * 512 + (((n & 15) | (kq & 16)) << 4) + (kq & 15);
    *reinterpret_cast<v4bf*>(d) = pk;
  }
}

// B fragment 32x16 directly from row-major fp32 global (small GEMMs).
DEV v16bf load_b16_gf32(const float* base, int stride, int lane) {
  const int col = lane & 15;
  const int kb  = (lane >> 4) << 4;
  v16bf b;
#pragma unroll
  for (int j = 0; j < 16; ++j) b[j] = (__bf16)base[(size_t)(kb + j) * stride + col];
  return b;
}

#if USE_TDM
DEV unsigned lds_offset(const void* p) {
  return (unsigned)(unsigned long long)(__attribute__((address_space(3))) const char*)p;
}

// Issue one TDM 2-D tile load (bf16 / 2-byte elements) global -> LDS.
// D# per CDNA5 ISA 8.3/8.4: count=1, type=2, data_size=1 (2B), 2-D tile.
// This toolchain exposes the 6-arg builtin:
//   (uint32x4 g0, int32x8 g1, int32x4 g2, int32x4 g3, int32x8 extra, int cpol)
DEV void tdm_load_2d_bf16(unsigned lds_byte_off, const void* gptr,
                          unsigned tileW, unsigned tileH,
                          unsigned rowStride, unsigned tensorW, unsigned tensorH) {
  unsigned long long ga = (unsigned long long)(size_t)gptr;
  v4u g0;
  g0[0] = 1u;                                        // count=1 (valid user D#)
  g0[1] = lds_byte_off;                              // lds_addr
  g0[2] = (unsigned)(ga & 0xffffffffu);              // global_addr[31:0]
  g0[3] = (unsigned)((ga >> 32) & 0x01ffffffu) | 0x80000000u;  // [56:32] | type=2
  v8i g1;
  g1[0] = (int)(1u << 16);                           // data_size=1 (2 bytes)
  g1[1] = (int)((tensorW & 0xffffu) << 16);          // tensor_dim0[15:0]
  g1[2] = (int)((tensorW >> 16) | ((tensorH & 0xffffu) << 16));
  g1[3] = (int)((tensorH >> 16) | (tileW << 16));    // tile_dim0
  g1[4] = (int)(tileH & 0xffffu);                    // tile_dim1 (tile_dim2=0)
  g1[5] = (int)rowStride;                            // tensor_dim0_stride[31:0]
  g1[6] = 0;                                         // stride hi, dim1_stride lo
  g1[7] = 0;
  v4i z4 = {0, 0, 0, 0};
  v8i z8 = {0, 0, 0, 0, 0, 0, 0, 0};
  __builtin_amdgcn_tensor_load_to_lds(g0, g1, z4, z4, z8, 0);
}
#endif

// ---------------------------------------------------------------------------
// Kernel 0: bf16 pre-convert of x (feeds TDM byte-move staging in enc1).
// ---------------------------------------------------------------------------
__global__ __launch_bounds__(256) void cvt_kernel(const float* __restrict__ src,
                                                  __bf16* __restrict__ dst, int n4) {
  int i = blockIdx.x * 256 + threadIdx.x;
  if (i < n4) {
    const float4 v = reinterpret_cast<const float4*>(src)[i];
    v4bf o = {(__bf16)v.x, (__bf16)v.y, (__bf16)v.z, (__bf16)v.w};
    reinterpret_cast<v4bf*>(dst)[i] = o;
  }
}

// ---------------------------------------------------------------------------
// Kernel 1: h = relu(xb @ enc_w1 + enc_b1), bf16 out (M1 x D).
// Block tile 64x128, 8 waves = 4(M) x 2(N). A tile staged by TDM (wave 0)
// while the other waves stage the swizzled B tile.
// ---------------------------------------------------------------------------
__global__ __launch_bounds__(256) void enc1_kernel(
    const __bf16* __restrict__ xb, const float* __restrict__ w1,
    const float* __restrict__ b1, __bf16* __restrict__ h) {
  __shared__ alignas(32) __bf16 sA[64 * 32];   // 4 KB, row-major
  __shared__ alignas(32) __bf16 sB[8 * 512];   // 8 KB, fragment-major
  const int tid = threadIdx.x, lane = tid & 31, wv = tid >> 5;
  const int mg = wv >> 1, ng = wv & 1;
  const int row0 = blockIdx.x * 64, col0 = blockIdx.y * 128;
  v8f acc[4] = {};

  for (int k0 = 0; k0 < Dv; k0 += 32) {
    if (k0 + 32 < Dv)  // gfx1250 global_prefetch_b8 on next W1 k-slab
      __builtin_prefetch(&w1[(size_t)(k0 + 32 + (tid >> 3)) * Dv + col0 + ((tid & 7) << 4)], 0, 1);
#if USE_TDM
    if (wv == 0) {     // one TDM DMA moves the whole 64x32 bf16 A tile
      tdm_load_2d_bf16(lds_offset(sA), xb + (size_t)row0 * Dv + k0,
                       /*tileW=*/32, /*tileH=*/64, /*stride=*/Dv,
                       /*tensorW=*/Dv, /*tensorH=*/64);
      __builtin_amdgcn_s_wait_tensorcnt(0);
    }
#else
#pragma unroll
    for (int i = tid; i < 64 * 4; i += 256) {  // 8-elem chunks
      const int r = i >> 2, s = (i & 3) << 3;
      *reinterpret_cast<v8bf*>(&sA[r * 32 + s]) =
          *reinterpret_cast<const v8bf*>(&xb[(size_t)(row0 + r) * Dv + k0 + s]);
    }
#endif
    stage_b_sw<128>(sB, w1 + (size_t)k0 * Dv + col0, Dv, tid);
    __syncthreads();
    v16bf af = load_a16(&sA[mg * 16 * 32], 32, lane);
#pragma unroll
    for (int j = 0; j < 4; ++j)
      acc[j] = wmma_bf16(af, load_b_sw(sB, ng * 4 + j, lane), acc[j]);
    __syncthreads();
  }
#pragma unroll
  for (int j = 0; j < 4; ++j)
#pragma unroll
    for (int r = 0; r < 8; ++r) {
      int row = row0 + mg * 16 + r + ((lane >> 4) << 3);
      int col = col0 + ng * 64 + j * 16 + (lane & 15);
      h[(size_t)row * Dv + col] = (__bf16)fmaxf(acc[j][r] + b1[col], 0.0f);
    }
}

// ---------------------------------------------------------------------------
// Kernel 2: probs = softmax(h @ enc_w2 + enc_b2) (TAU=1), bf16 out (M1 x 32).
// ---------------------------------------------------------------------------
__global__ __launch_bounds__(256) void enc2_softmax_kernel(
    const __bf16* __restrict__ h, const float* __restrict__ w2,
    const float* __restrict__ b2, __bf16* __restrict__ probs) {
  __shared__ alignas(32) __bf16 sB[2 * 512];   // 2 KB fragment-major
  __shared__ float sLog[128][32];              // 16 KB
  const int tid = threadIdx.x, lane = tid & 31, wv = tid >> 5;
  const int row0 = blockIdx.x * 128;
  const int arow = row0 + wv * 16 + (lane & 15);
  const int ksel = (lane >> 4) << 4;
  v8f acc[2] = {};

  for (int k0 = 0; k0 < Dv; k0 += 32) {
    stage_b_sw<32>(sB, w2 + (size_t)k0 * Sv, Sv, tid);
    __syncthreads();
    const __bf16* ap = h + (size_t)arow * Dv + k0 + ksel;
    v8bf lo = *reinterpret_cast<const v8bf*>(ap);
    v8bf hi = *reinterpret_cast<const v8bf*>(ap + 8);
    v16bf af = __builtin_shufflevector(lo, hi, 0, 1, 2, 3, 4, 5, 6, 7,
                                       8, 9, 10, 11, 12, 13, 14, 15);
#pragma unroll
    for (int n = 0; n < 2; ++n)
      acc[n] = wmma_bf16(af, load_b_sw(sB, n, lane), acc[n]);
    __syncthreads();
  }
#pragma unroll
  for (int n = 0; n < 2; ++n)
#pragma unroll
    for (int r = 0; r < 8; ++r) {
      int lr = wv * 16 + r + ((lane >> 4) << 3);
      int c = n * 16 + (lane & 15);
      sLog[lr][c] = acc[n][r] + b2[c];  // /TAU with TAU==1
    }
  __syncthreads();
  if (tid < 128) {
    float m = -3.4e38f;
#pragma unroll
    for (int j = 0; j < 32; ++j) m = fmaxf(m, sLog[tid][j]);
    float e[32], s = 0.0f;
#pragma unroll
    for (int j = 0; j < 32; ++j) { e[j] = __expf(sLog[tid][j] - m); s += e[j]; }
    float inv = 1.0f / s;
#pragma unroll
    for (int j = 0; j < 32; ++j)
      probs[(size_t)(row0 + tid) * Sv + j] = (__bf16)(e[j] * inv);
  }
}

// ---------------------------------------------------------------------------
// Kernel 3: trans = softmax32(relu(cur @ tr_w1 + b1) @ tr_w2 + b2)
// ---------------------------------------------------------------------------
__global__ __launch_bounds__(256) void trans_kernel(
    const __bf16* __restrict__ probs, const float* __restrict__ w1,
    const float* __restrict__ b1, const float* __restrict__ w2,
    const float* __restrict__ b2, float* __restrict__ trans) {
  __shared__ alignas(32) __bf16 sA[16 * 32];    // 1 KB row-major
  __shared__ alignas(32) __bf16 sW1[4 * 512];   // 4 KB fragment-major
  __shared__ alignas(32) __bf16 sTh[16 * 64];   // 2 KB row-major
  __shared__ alignas(32) __bf16 sTl[16 * 1024]; // 32 KB
  const int tid = threadIdx.x, lane = tid & 31, wv = tid >> 5;
  const int r0 = blockIdx.x * 16;

  for (int i = tid; i < 16 * 32; i += 256) {
    int lr = i >> 5, c = i & 31;
    int gr = r0 + lr;
    int bi = gr / 1023;                    // skip last row of each batch
    int src = bi * 1024 + (gr - bi * 1023);
    sA[lr * 32 + c] = probs[(size_t)src * Sv + c];
  }
  stage_b_sw<64>(sW1, w1, 64, tid);
  __syncthreads();

  if (wv < 4) {  // gemm1: 16 x 64, K=32, one tile per wave
    v16bf af = load_a16(sA, 32, lane);
    v8f acc = {};
    acc = wmma_bf16(af, load_b_sw(sW1, wv, lane), acc);
#pragma unroll
    for (int r = 0; r < 8; ++r) {
      int row = r + ((lane >> 4) << 3);
      int c = wv * 16 + (lane & 15);
      sTh[row * 64 + c] = (__bf16)fmaxf(acc[r] + b1[c], 0.0f);
    }
  }
  __syncthreads();

#pragma unroll 1
  for (int t = 0; t < 8; ++t) {  // gemm2: 16 x 1024, K=64
    int nsub = wv * 8 + t;
    int c = nsub * 16 + (lane & 15);
    v8f acc = {};
#pragma unroll
    for (int ks = 0; ks < 2; ++ks) {
      v16bf af = load_a16(sTh + ks * 32, 64, lane);
      v16bf bf = load_b16_gf32(w2 + (size_t)ks * 32 * 1024 + nsub * 16, 1024, lane);
      acc = wmma_bf16(af, bf, acc);
    }
#pragma unroll
    for (int r = 0; r < 8; ++r)
      sTl[(r + ((lane >> 4) << 3)) * 1024 + c] = (__bf16)(acc[r] + b2[c]);
  }
  __syncthreads();

  for (int jj = tid; jj < 16 * 32; jj += 256) {  // 32-wide softmax groups
    int row = jj >> 5, g = jj & 31;
    const __bf16* src = &sTl[row * 1024 + g * 32];
    float m = -3.4e38f;
#pragma unroll
    for (int j = 0; j < 32; ++j) m = fmaxf(m, (float)src[j]);
    float e[32], s = 0.0f;
#pragma unroll
    for (int j = 0; j < 32; ++j) { e[j] = __expf((float)src[j] - m); s += e[j]; }
    float inv = 1.0f / s;
    size_t base = (((size_t)(r0 + row)) * 32 + g) * 32;
#pragma unroll
    for (int j = 0; j < 32; ++j) trans[base + j] = e[j] * inv;
  }
}

// ---------------------------------------------------------------------------
// Kernel 4: markov = relu(probs @ dec_w1 + b1) @ dec_w2 + b2
// Block: 16 rows x 256 cols; dh (16x1024 bf16) lives in LDS.
// ---------------------------------------------------------------------------
__global__ __launch_bounds__(256) void dec_kernel(
    const __bf16* __restrict__ probs, const float* __restrict__ w1,
    const float* __restrict__ b1, const float* __restrict__ w2,
    const float* __restrict__ b2, float* __restrict__ out) {
  __shared__ alignas(32) __bf16 sP[16 * 32];     // 1 KB row-major
  __shared__ alignas(32) __bf16 sDh[16 * 1024];  // 32 KB row-major
  __shared__ alignas(32) __bf16 sB[16 * 512];    // 16 KB fragment-major
  const int tid = threadIdx.x, lane = tid & 31, wv = tid >> 5;
  const int r0 = blockIdx.x * 16, c0 = blockIdx.y * 256;

  for (int i = tid; i < 16 * 32; i += 256)
    sP[(i >> 5) * 32 + (i & 31)] = probs[(size_t)(r0 + (i >> 5)) * Sv + (i & 31)];
  __syncthreads();

  {  // gemm1: dh = relu(probs @ dec_w1 + b1), 16 x 1024, K=32
    v16bf af = load_a16(sP, 32, lane);
#pragma unroll 1
    for (int t = 0; t < 8; ++t) {
      int nsub = wv + 8 * t;
      int c = nsub * 16 + (lane & 15);
      v16bf bf = load_b16_gf32(w1 + nsub * 16, 1024, lane);
      v8f acc = {};
      acc = wmma_bf16(af, bf, acc);
#pragma unroll
      for (int r = 0; r < 8; ++r)
        sDh[(r + ((lane >> 4) << 3)) * 1024 + c] = (__bf16)fmaxf(acc[r] + b1[c], 0.0f);
    }
  }
  __syncthreads();

  v8f acc2[2] = {};
#pragma unroll 1
  for (int ks = 0; ks < 32; ++ks) {  // gemm2: K = 1024 in 32 steps
    stage_b_sw<256>(sB, w2 + (size_t)(ks * 32) * Dv + c0, Dv, tid);
    __syncthreads();
    v16bf af = load_a16(sDh + ks * 32, 1024, lane);
#pragma unroll
    for (int n = 0; n < 2; ++n)
      acc2[n] = wmma_bf16(af, load_b_sw(sB, wv * 2 + n, lane), acc2[n]);
    __syncthreads();
  }
#pragma unroll
  for (int n = 0; n < 2; ++n)
#pragma unroll
    for (int r = 0; r < 8; ++r) {
      int row = r0 + r + ((lane >> 4) << 3);
      int c = c0 + (wv * 2 + n) * 16 + (lane & 15);
      out[(size_t)row * 1024 + c] = acc2[n][r] + b2[c];
    }
}

// ---------------------------------------------------------------------------
// Host launcher
// ---------------------------------------------------------------------------
extern "C" void kernel_launch(void* const* d_in, const int* in_sizes, int n_in,
                              void* d_out, int out_size, void* d_ws, size_t ws_size,
                              hipStream_t stream) {
  (void)in_sizes; (void)n_in; (void)out_size; (void)ws_size;
  const float* x      = (const float*)d_in[0];
  const float* enc_w1 = (const float*)d_in[1];
  const float* enc_b1 = (const float*)d_in[2];
  const float* enc_w2 = (const float*)d_in[3];
  const float* enc_b2 = (const float*)d_in[4];
  const float* tr_w1  = (const float*)d_in[5];
  const float* tr_b1  = (const float*)d_in[6];
  const float* tr_w2  = (const float*)d_in[7];
  const float* tr_b2  = (const float*)d_in[8];
  const float* dec_w1 = (const float*)d_in[9];
  const float* dec_b1 = (const float*)d_in[10];
  const float* dec_w2 = (const float*)d_in[11];
  const float* dec_b2 = (const float*)d_in[12];

  float* markov = (float*)d_out;                    // B*L*D floats
  float* trans  = (float*)d_out + (size_t)M1 * Dv;  // B*(L-1)*S*S floats
  // Scratch overlays (all dependencies are stream-ordered):
  //   xb (bf16 x, 67 MB) -> front of trans region, consumed by enc1, then
  //                         overwritten by trans_kernel.
  //   h  (bf16, 67 MB)   -> front of markov region, consumed by enc2, then
  //                         overwritten by dec_kernel.
  //   probs (bf16, 2 MB) -> workspace.
  __bf16* xb    = (__bf16*)trans;
  __bf16* h     = (__bf16*)d_out;
  __bf16* probs = (__bf16*)d_ws;

  cvt_kernel<<<dim3((M1 / 4) * (Dv / 256)), 256, 0, stream>>>(x, xb, M1 * Dv / 4);
  enc1_kernel<<<dim3(M1 / 64, Dv / 128), 256, 0, stream>>>(xb, enc_w1, enc_b1, h);
  enc2_softmax_kernel<<<dim3(M1 / 128), 256, 0, stream>>>(h, enc_w2, enc_b2, probs);
  trans_kernel<<<dim3(M2 / 16), 256, 0, stream>>>(probs, tr_w1, tr_b1, tr_w2, tr_b2, trans);
  dec_kernel<<<dim3(M1 / 16, 4), 256, 0, stream>>>(probs, dec_w1, dec_b1, dec_w2, dec_b2, markov);
}
